// HungarianMatcher_85890755985627
// MI455X (gfx1250) — compile-verified, matrix operand order checked by reading, not run on hardware
//
#include <hip/hip_runtime.h>
#include <hip/hip_bf16.h>
#include <stdint.h>

typedef __attribute__((ext_vector_type(16))) _Float16 v16h;
typedef __attribute__((ext_vector_type(8)))  _Float16 v8h;
typedef __attribute__((ext_vector_type(8)))  float    v8f;

#define B_   8
#define N_   256
#define C_   8192
#define R_   8
#define NC_  25
#define KT_  256                 // k-tile size
#define NKT_ (C_ / KT_)          // 32 k-tiles
#define ASTRIDE 264              // 256 + 8 halves pad: rows stay 16B aligned, banks spread

union Frag { v16h v; v8h h[2]; uint32_t u[8]; };
union H8   { v8h v; _Float16 h[8]; };

__global__ __launch_bounds__(512)
void hungarian_cost_kernel(const float* __restrict__ mask_pred,
                           const float* __restrict__ class_probs,
                           const float* __restrict__ reg_pred,
                           const float* __restrict__ reg_tgt,
                           const int*   __restrict__ object_class,
                           const int*   __restrict__ mask_tgt,
                           float* __restrict__ out)
{
    // double-buffered f16 A tiles: [buf][row*ASTRIDE + col]
    __shared__ __align__(16) _Float16 ApB[2][16*ASTRIDE];  // sigmoid(x)
    __shared__ __align__(16) _Float16 AxB[2][16*ASTRIDE];  // x
    __shared__ __align__(16) _Float16 AfB[2][16*ASTRIDE];  // (1-p)^2*pos - p^2*neg
    __shared__ float rt_lds[N_*R_];
    __shared__ float rp_lds[16*R_];
    __shared__ float cp_lds[16*NC_];
    __shared__ int   oc_lds[NC_];
    __shared__ int   valid_lds[N_];
    __shared__ float sp_lds[16], sn_lds[16], sf_lds[16];
    __shared__ float st_lds[512];
    __shared__ int   len_lds;

    const int tid  = threadIdx.x;
    const int wave = tid >> 5;          // 16 waves, one 16-wide m-tile each
    const int lane = tid & 31;
    const int b    = blockIdx.y;
    const int n0   = blockIdx.x * 16;
    const int m0   = wave * 16;

    if (tid < 16) { sp_lds[tid] = 0.f; sn_lds[tid] = 0.f; sf_lds[tid] = 0.f; }
    if (tid == 0) len_lds = 0;
    __syncthreads();
    if (tid < N_) {
        int cls = object_class[b*N_ + tid];
        int v = (cls != NC_) ? 1 : 0;
        valid_lds[tid] = v;
        atomicAdd(&len_lds, v);
        if (tid < NC_) oc_lds[tid] = cls;
    }
    __syncthreads();
    for (int i = tid; i < N_*R_; i += 512) {
        int m = i >> 3;
        rt_lds[i] = reg_tgt[(size_t)(b*N_ + m)*R_ + (i & 7)] * (float)valid_lds[m];
    }
    for (int i = tid; i < 16*R_; i += 512)
        rp_lds[i] = reg_pred[(size_t)(b*N_ + n0 + (i >> 3))*R_ + (i & 7)];
    for (int i = tid; i < 16*NC_; i += 512)
        cp_lds[i] = class_probs[(size_t)(b*N_ + n0 + i/NC_)*NC_ + (i % NC_)];

    // preprocess mapping: 16 rows x 256 cols, 8 contiguous cols per thread
    const int prow = tid >> 5;
    const int pcol = (tid & 31) * 8;
    const float* mp_row = mask_pred + (size_t)(b*N_ + n0 + prow) * C_;

    // WMMA fragment mapping (ISA 16-bit A 16x32 layout):
    // lanes 0-15: row=lane,    K chunks [0..7] and [16..23]
    // lanes16-31: row=lane-16, K chunks [8..15] and [24..31]
    const int frow = lane & 15;
    const int fhal = lane >> 4;
    const int* mt_row = mask_tgt + (size_t)(b*N_ + m0 + frow) * C_;

    v8f cP = {}; v8f cX = {}; v8f cF = {};
    int st_part = 0;

    // elementwise preprocessing of one 16x256 mask_pred tile into one LDS buffer
    auto preprocess = [&](int k0, _Float16* Ap, _Float16* Ax, _Float16* Af) {
        float4 xa = *(const float4*)(mp_row + k0 + pcol);
        float4 xb = *(const float4*)(mp_row + k0 + pcol + 4);
        float xs[8] = {xa.x, xa.y, xa.z, xa.w, xb.x, xb.y, xb.z, xb.w};
        float lsp = 0.f, lsn = 0.f, lsf = 0.f;
        H8 hp, hx, hf;
        #pragma unroll
        for (int i = 0; i < 8; ++i) {
            float x    = xs[i];
            float e    = __expf(-fabsf(x));
            float inv  = 1.f / (1.f + e);
            float p    = (x >= 0.f) ? inv : e * inv;      // sigmoid
            float sneg = fmaxf(x, 0.f) + log1pf(e);       // softplus(x)
            float pos  = sneg - x;                        // softplus(-x)
            float omp  = 1.f - p;
            float fneg = p * p * sneg;
            float fdif = omp * omp * pos - fneg;          // fpos - fneg
            lsp += p; lsn += sneg; lsf += fneg;
            hp.h[i] = (_Float16)p;
            hx.h[i] = (_Float16)x;
            hf.h[i] = (_Float16)fdif;
        }
        *(v8h*)(Ap + prow*ASTRIDE + pcol) = hp.v;
        *(v8h*)(Ax + prow*ASTRIDE + pcol) = hx.v;
        *(v8h*)(Af + prow*ASTRIDE + pcol) = hf.v;
        atomicAdd(&sp_lds[prow], lsp);
        atomicAdd(&sn_lds[prow], lsn);
        atomicAdd(&sf_lds[prow], lsf);
    };

    // prologue: stage tile 0 into buffer 0
    preprocess(0, ApB[0], AxB[0], AfB[0]);
    __syncthreads();

    // software-pipelined main loop: one barrier per iteration; preprocess of
    // tile kt+1 (VALU+global loads) overlaps the WMMA stage of tile kt.
    for (int kt = 0; kt < NKT_; ++kt) {
        const int cur = kt & 1;
        if (kt + 1 < NKT_)
            preprocess((kt + 1) * KT_, ApB[1 - cur], AxB[1 - cur], AfB[1 - cur]);

        const int k0 = kt * KT_;
        const _Float16* Ap = ApB[cur];
        const _Float16* Ax = AxB[cur];
        const _Float16* Af = AfB[cur];

        #pragma unroll
        for (int ks = 0; ks < 8; ++ks) {
            const int koff = ks*32 + fhal*8;
            // B fragment: binary mask target, int32 0/1 -> packed f16 pairs
            const int* tp = mt_row + k0 + koff;
            Frag fb;
            {
                int4 a0 = *(const int4*)(tp);
                int4 a1 = *(const int4*)(tp + 4);
                int4 b0 = *(const int4*)(tp + 16);
                int4 b1 = *(const int4*)(tp + 20);
                st_part += a0.x+a0.y+a0.z+a0.w + a1.x+a1.y+a1.z+a1.w
                         + b0.x+b0.y+b0.z+b0.w + b1.x+b1.y+b1.z+b1.w;
                fb.u[0] = (uint32_t)(a0.x | (a0.y << 16)) * 0x3C00u;
                fb.u[1] = (uint32_t)(a0.z | (a0.w << 16)) * 0x3C00u;
                fb.u[2] = (uint32_t)(a1.x | (a1.y << 16)) * 0x3C00u;
                fb.u[3] = (uint32_t)(a1.z | (a1.w << 16)) * 0x3C00u;
                fb.u[4] = (uint32_t)(b0.x | (b0.y << 16)) * 0x3C00u;
                fb.u[5] = (uint32_t)(b0.z | (b0.w << 16)) * 0x3C00u;
                fb.u[6] = (uint32_t)(b1.x | (b1.y << 16)) * 0x3C00u;
                fb.u[7] = (uint32_t)(b1.z | (b1.w << 16)) * 0x3C00u;
            }
            const int aoff = frow*ASTRIDE + koff;
            Frag fa;
            fa.h[0] = *(const v8h*)(Ap + aoff);
            fa.h[1] = *(const v8h*)(Ap + aoff + 16);
            cP = __builtin_amdgcn_wmma_f32_16x16x32_f16(false, fa.v, false, fb.v,
                                                        (short)0, cP, false, false);
            fa.h[0] = *(const v8h*)(Ax + aoff);
            fa.h[1] = *(const v8h*)(Ax + aoff + 16);
            cX = __builtin_amdgcn_wmma_f32_16x16x32_f16(false, fa.v, false, fb.v,
                                                        (short)0, cX, false, false);
            fa.h[0] = *(const v8h*)(Af + aoff);
            fa.h[1] = *(const v8h*)(Af + aoff + 16);
            cF = __builtin_amdgcn_wmma_f32_16x16x32_f16(false, fa.v, false, fb.v,
                                                        (short)0, cF, false, false);
        }
        __syncthreads();   // buffer written this iter is read next iter; buffer
                           // read this iter is overwritten next iter
    }

    st_lds[tid] = (float)st_part;
    __syncthreads();

    // C/D f32 16x16 layout: VGPR j, lane l -> row = j + 8*(l>=16), col = l&15
    const int col  = lane & 15;
    const int m    = m0 + col;
    const float stm = st_lds[wave*32 + col] + st_lds[wave*32 + col + 16];
    const int len  = len_lds;
    const float* rtm = rt_lds + m*R_;

    #pragma unroll
    for (int j = 0; j < 8; ++j) {
        const int nl = j + ((lane >= 16) ? 8 : 0);
        float dice = 1.f - (2.f*cP[j] + 1.f) / (sp_lds[nl] + stm + 1.f);
        float ce   = (sn_lds[nl] - cX[j]) * (1.f / (float)C_);
        float fo   = (sf_lds[nl] + cF[j]) * (1.f / (float)C_);
        float cls  = 0.f;
        if (m < NC_) {
            int oc = oc_lds[m];
            if (oc != NC_) cls = -cp_lds[nl*NC_ + oc];
        }
        float mae = 0.f;
        const float* rpn = rp_lds + nl*R_;
        #pragma unroll
        for (int r = 0; r < R_; ++r) mae += fabsf(rpn[r] - rtm[r]);
        mae *= (1.f / (float)R_);
        float cost = 2.f*cls + 2.f*dice + 5.f*ce + 5.f*fo + mae;
        if (m >= len) cost = __builtin_nanf("");
        out[(size_t)(b*N_ + n0 + nl)*N_ + m] = cost;
    }
    // second tuple output: lengths[:, None], written once per batch
    if (blockIdx.x == 0 && tid == 0)
        out[(size_t)B_*N_*N_ + b] = (float)len;
}

extern "C" void kernel_launch(void* const* d_in, const int* in_sizes, int n_in,
                              void* d_out, int out_size, void* d_ws, size_t ws_size,
                              hipStream_t stream) {
    (void)in_sizes; (void)n_in; (void)out_size; (void)d_ws; (void)ws_size;
    dim3 grid(N_/16, B_);
    hungarian_cost_kernel<<<grid, dim3(512), 0, stream>>>(
        (const float*)d_in[0],   // mask_pred   (B,N,C) f32
        (const float*)d_in[1],   // class_probs (B,N,25) f32
        (const float*)d_in[2],   // reg_pred    (B,N,8) f32
        (const float*)d_in[3],   // reg_tgt     (B,N,8) f32
        (const int*)d_in[4],     // object_class(B,N) i32
        (const int*)d_in[5],     // mask_tgt    (B,N,C) i32
        (float*)d_out);
}
